// SparseSeqKVAttention_V7_17669495456354
// MI455X (gfx1250) — matrix operand order repaired and checked by reading, not compiled, not run on hardware
//
#include <hip/hip_runtime.h>
#include <hip/hip_bf16.h>
#include <math.h>
#include <stdint.h>

typedef _Float16 half_t;
typedef __attribute__((ext_vector_type(16))) _Float16     v16h;
typedef __attribute__((ext_vector_type(8)))  float        v8f;
typedef __attribute__((ext_vector_type(4)))  unsigned int v4u;
typedef __attribute__((ext_vector_type(4)))  int          v4i;
typedef __attribute__((ext_vector_type(8)))  int          v8i;

#define B_    4
#define T_    512
#define D_    512
#define E_    512
#define N_    4096
#define H_    8
#define HD_   36
#define NH_   512      // N_/H_
#define TTOT_ 1536     // 3*T_
#define TOPK_ 16
#define KPAD_ 64       // padded K-dim for HD=36 operands
#define HHD_  288      // H_*HD_
#define NTOK_ALL_ 22528 // 3*B_*T_ + B_*N_

#if __has_builtin(__builtin_amdgcn_tensor_load_to_lds)
#define HAVE_TDM 1
#else
#define HAVE_TDM 0
#endif

// ---------------- WMMA helpers (CDNA5 16x16x32 f16 -> f32) ----------------

__device__ __forceinline__ v8f wmma16(v16h a, v16h b, v8f c) {
  return __builtin_amdgcn_wmma_f32_16x16x32_f16(false, a, false, b, (short)0, c,
                                                false, false);
}

// A fragment: 16x32 f16 tile, row-major source, row stride lda (halves).
// Layout: lanes 0-15 hold K=0..15 (elts 0-15),
// lanes 16-31 hold K=16..31, row M = lane&15.
__device__ __forceinline__ v16h load_a16(const half_t* A, int lda, int lane) {
  int r = lane & 15, kg = lane >> 4;
  const half_t* p = A + (size_t)r * lda + kg * 16;
  v16h a;
#pragma unroll
  for (int j = 0; j < 16; ++j) { a[j] = p[j]; }
  return a;
}

// B fragment (32x16, K x N) for computing A * M^T where M is an N x K
// row-major f16 matrix. Lane n<16 holds K=0..15 of column n, n+16 K=16..31.
__device__ __forceinline__ v16h load_bT16(const half_t* M, int ldm, int lane) {
  int c = lane & 15, kg = lane >> 4;
  const half_t* p = M + (size_t)c * ldm + kg * 16;
  v16h b;
#pragma unroll
  for (int e = 0; e < 16; ++e) b[e] = p[e];
  return b;
}

// ---------------- K0a: f32 -> f16 weight preconversion --------------------

__global__ void cvt_kernel(const float* __restrict__ s, half_t* __restrict__ d,
                           int n) {
  int i = blockIdx.x * 256 + threadIdx.x;
  if (i < n) d[i] = (half_t)s[i];
}

// ---------------- K0b: modulation  mod[l][b][j] = silu(emb[b])@W_l^T + b_l

__global__ void mod_kernel(const float* __restrict__ emb,
                           const float* __restrict__ adaln_w,
                           const float* __restrict__ adaln_b,
                           const float* __restrict__ xfn_w,
                           const float* __restrict__ xfn_b,
                           float* __restrict__ mod) {
  int idx = blockIdx.x * blockDim.x + threadIdx.x;   // 0 .. 4*B_*1024-1
  int l = idx >> 12;                                  // layer 0..3
  int r = idx & 4095;
  int b = r >> 10;
  int j = r & 1023;
  const float* w = (l < 3) ? adaln_w + ((size_t)l * 2 * D_ + j) * E_
                           : xfn_w + (size_t)j * E_;
  float acc = (l < 3) ? adaln_b[l * 2 * D_ + j] : xfn_b[j];
  const float* e = emb + (size_t)b * E_;
#pragma unroll 4
  for (int k = 0; k < E_; ++k) {
    float x = e[k];
    acc += (x / (1.0f + expf(-x))) * w[k];
  }
  mod[idx] = acc;
}

// ---------------- K1: AdaLN -> f16 token matrix (wave per token) ----------

__global__ void adaln_kernel(const float* __restrict__ x1,
                             const float* __restrict__ x2,
                             const float* __restrict__ x3,
                             const float* __restrict__ xf,
                             const float* __restrict__ mod,
                             half_t* __restrict__ xn) {
  int token = blockIdx.x * 8 + (threadIdx.x >> 5);
  int lane  = threadIdx.x & 31;
  int l, b;
  const float* src;
  if (token < 6144) {                 // branch tokens: [b*512 + t]
    l = token >> 11;
    int r = token & 2047;
    b = r >> 9;
    const float* base = (l == 0) ? x1 : (l == 1) ? x2 : x3;
    src = base + (size_t)r * D_;
  } else {                            // xf tokens
    l = 3;
    int r = token - 6144;
    b = r >> 12;
    src = xf + (size_t)r * D_;
  }
  float v[16];
  float s = 0.f, s2 = 0.f;
#pragma unroll
  for (int j = 0; j < 16; ++j) {
    v[j] = src[lane + 32 * j];
    s += v[j];
    s2 += v[j] * v[j];
  }
#pragma unroll
  for (int off = 16; off >= 1; off >>= 1) {
    s  += __shfl_xor(s, off, 32);
    s2 += __shfl_xor(s2, off, 32);
  }
  float mean = s * (1.0f / D_);
  float var  = s2 * (1.0f / D_) - mean * mean;
  float rstd = rsqrtf(var + 1e-6f);
  const float* mrow = mod + ((size_t)(l * B_ + b)) * (2 * D_);
  half_t* dst = xn + (size_t)token * D_;
#pragma unroll
  for (int j = 0; j < 16; ++j) {
    int d = lane + 32 * j;
    float h = (v[j] - mean) * rstd;
    dst[d] = (half_t)(h * (1.0f + mrow[d]) + mrow[D_ + d]);
  }
}

// ---------------- K2: Q projection -> Qf16[b][h][t'][64] ------------------

__global__ void qproj_kernel(const half_t* __restrict__ xn,
                             const half_t* __restrict__ qw16,
                             const float* __restrict__ q_b,
                             half_t* __restrict__ Qf) {
  int wave = blockIdx.x * 8 + (threadIdx.x >> 5);   // 384 mtiles * 18 ntiles
  int lane = threadIdx.x & 31;
  int mt = wave / 18, nt = wave % 18;
  int row0 = mt * 16;                 // global token row, 0..6143
  int branch = row0 >> 11;
  int n0 = nt * 16;
  const half_t* A = xn + (size_t)row0 * D_;
  const half_t* W = qw16 + (size_t)(branch * HHD_ + n0) * D_;
  v8f acc = {};
#pragma unroll 4
  for (int kt = 0; kt < D_ / 32; ++kt) {
    v16h a = load_a16(A + kt * 32, D_, lane);
    v16h b = load_bT16(W + kt * 32, D_, lane);
    acc = wmma16(a, b, acc);
  }
  int col = n0 + (lane & 15);
  int h = col / HD_, k = col - h * HD_;
  int bb = (row0 & 2047) >> 9;
  int tb = branch * T_ + (row0 & 511);
  float bias = q_b[branch * HHD_ + col];
#pragma unroll
  for (int r = 0; r < 8; ++r) {
    int mloc = 8 * (lane >> 4) + r;
    Qf[(((size_t)(bb * H_ + h) * TTOT_) + tb + mloc) * KPAD_ + k] =
        (half_t)(acc[r] + bias);
  }
}

// ---------------- K3: K & V projections (shared A fragment) ---------------

__global__ void kv_kernel(const half_t* __restrict__ xnf,   // xf section
                          const half_t* __restrict__ kw16,
                          const float* __restrict__ k_b,
                          const half_t* __restrict__ vw16,
                          const float* __restrict__ v_b,
                          half_t* __restrict__ Kf,
                          float* __restrict__ Vf) {
  int wave = blockIdx.x * 8 + (threadIdx.x >> 5);   // 1024 mtiles * 3 ntiles
  int lane = threadIdx.x & 31;
  int mt = wave / 3, nt = wave % 3;
  int row0 = mt * 16;                               // 0..16383
  int b = row0 >> 12;
  int h = (row0 & 4095) >> 9;
  int n0 = nt * 16;
  int col = n0 + (lane & 15);
  bool valid = col < HD_;
  int kg = lane >> 4;
  const half_t* A = xnf + (size_t)row0 * D_;
  const half_t* Wk = kw16 + (size_t)(h * HD_ + (valid ? col : 0)) * D_;
  const half_t* Wv = vw16 + (size_t)(h * HD_ + (valid ? col : 0)) * D_;
  v8f accK = {}, accV = {};
#pragma unroll 4
  for (int kt = 0; kt < D_ / 32; ++kt) {
    v16h a = load_a16(A + kt * 32, D_, lane);
    v16h bk, bv;
#pragma unroll
    for (int e = 0; e < 16; ++e) {
      bk[e] = valid ? Wk[kt * 32 + kg * 16 + e] : (half_t)0;
      bv[e] = valid ? Wv[kt * 32 + kg * 16 + e] : (half_t)0;
    }
    accK = wmma16(a, bk, accK);
    accV = wmma16(a, bv, accV);
  }
  if (valid) {
    float kb = k_b[h * HD_ + col];
    float vb = v_b[h * HD_ + col];
    int nloc0 = row0 & 511;
#pragma unroll
    for (int r = 0; r < 8; ++r) {
      int nloc = nloc0 + 8 * kg + r;
      size_t base = (size_t)(b * H_ + h) * NH_ + nloc;
      Kf[base * KPAD_ + col] = (half_t)(accK[r] + kb);
      Vf[base * HD_ + col]   = accV[r] + vb;
    }
  }
}

// ---------------- K4: fused logits + gumbel + top16 softmax + sparse AV ---

__global__ void attn_kernel(const half_t* __restrict__ Qf,
                            const half_t* __restrict__ Kf,
                            const float* __restrict__ Vf,
                            const float* __restrict__ u_noise,
                            half_t* __restrict__ attn) {
  __shared__ float lds[16][NH_];
  __shared__ __align__(16) half_t ldsQ[16 * KPAD_];   // staged Q tile (2KB)
  int mt = blockIdx.x;           // 0..95
  int h  = blockIdx.y;           // 0..7
  int b  = blockIdx.z;           // 0..3
  int tid = threadIdx.x;
  int w = tid >> 5, lane = tid & 31;
  int t0 = mt * 16;
  size_t bh = (size_t)(b * H_ + h);
  const half_t* Qbase = Qf + (bh * TTOT_ + t0) * KPAD_;
  const half_t* Kbase = Kf + bh * NH_ * KPAD_;
  const float*  Vbase = Vf + bh * NH_ * HD_;
  const float*  Ubase = u_noise + (bh * TTOT_ + t0) * NH_;

  // stage the 16x64 Q tile into LDS via the Tensor Data Mover (wave 0 only)
#if HAVE_TDM
  if (tid < 32) {
    unsigned long long ga = (unsigned long long)(size_t)Qbase;
    unsigned int lo = (unsigned int)(size_t)(&ldsQ[0]);
    v4u g0 = {};
    g0[0] = 1u;                                          // count=1, user mode
    g0[1] = lo;                                          // lds_addr (bytes)
    g0[2] = (unsigned int)ga;                            // global_addr[31:0]
    g0[3] = (unsigned int)((ga >> 32) & 0x01FFFFFFull)   // global_addr[56:32]
            | 0x80000000u;                               // type = 2 (image)
    v8i g1 = {};
    g1[0] = (int)(1u << 16);    // data_size = 2 bytes
    g1[1] = (int)(64u << 16);   // tensor_dim0 = 64
    g1[2] = (int)(16u << 16);   // tensor_dim1 = 16
    g1[3] = (int)(64u << 16);   // tile_dim0 = 64
    g1[4] = 16;                 // tile_dim1 = 16 (tile_dim2 = 0)
    g1[5] = 64;                 // tensor_dim0_stride = 64
    g1[6] = 0;
    g1[7] = 0;
    v4i gz = {};
#if defined(__clang_major__) && __clang_major__ >= 23
    v8i gz8 = {};
    __builtin_amdgcn_tensor_load_to_lds(g0, g1, gz, gz, gz8, 0);
#else
    __builtin_amdgcn_tensor_load_to_lds(g0, g1, gz, gz, 0);
#endif
    __builtin_amdgcn_s_wait_tensorcnt(0);
  }
#else
  if (tid < 128) ((int4*)ldsQ)[tid] = ((const int4*)Qbase)[tid];
#endif
  // prefetch the 32KB gumbel-noise tile while logits WMMAs run
  if (tid < 128) __builtin_prefetch(Ubase + (size_t)tid * 64, 0, 0);
  __syncthreads();

  // phase 1: logits tile 16 x 512 via WMMA, + gumbel, into LDS
#pragma unroll
  for (int q = 0; q < 4; ++q) {
    int nt = w * 4 + q;
    int n0 = nt * 16;
    v8f acc = {};
#pragma unroll
    for (int kt = 0; kt < KPAD_ / 32; ++kt) {
      v16h a = load_a16(&ldsQ[kt * 32], KPAD_, lane);
      v16h bm = load_bT16(Kbase + (size_t)n0 * KPAD_ + kt * 32, KPAD_, lane);
      acc = wmma16(a, bm, acc);
    }
    int col = n0 + (lane & 15);
#pragma unroll
    for (int r = 0; r < 8; ++r) {
      int mloc = 8 * (lane >> 4) + r;
      float u = Ubase[(size_t)mloc * NH_ + col];
      float g = -logf(-logf(u + 1e-9f) + 1e-9f);
      lds[mloc][col] = acc[r] * (1.0f / 6.0f) + g;   // 1/sqrt(36)
    }
  }
  __syncthreads();

  // phase 2: per row, iterative top-16 argmax + softmax + sparse AV
  for (int rr = 0; rr < 2; ++rr) {
    int mloc = w * 2 + rr;
    volatile float* row = &lds[mloc][0];
    float m1 = 0.f, denom = 0.f, acc0 = 0.f, acc1 = 0.f;
    for (int it = 0; it < TOPK_; ++it) {
      float bestv = -INFINITY;
      int bestn = 0x7fffffff;
#pragma unroll
      for (int j = 0; j < 16; ++j) {
        int n = lane + 32 * j;
        float val = row[n];
        if (val > bestv) { bestv = val; bestn = n; }
      }
#pragma unroll
      for (int off = 16; off >= 1; off >>= 1) {
        float ov = __shfl_xor(bestv, off, 32);
        int   on = __shfl_xor(bestn, off, 32);
        if (ov > bestv || (ov == bestv && on < bestn)) { bestv = ov; bestn = on; }
      }
      if (it == 0) m1 = bestv;
      float wgt = expf(bestv - m1);
      denom += wgt;
      const float* vp = Vbase + (size_t)bestn * HD_;
      acc0 += wgt * vp[lane];
      if (lane < HD_ - 32) acc1 += wgt * vp[32 + lane];
      if (lane == 0) row[bestn] = -INFINITY;
    }
    float inv = 1.0f / denom;
    half_t* dst = attn + ((size_t)b * TTOT_ + t0 + mloc) * HHD_ + h * HD_;
    dst[lane] = (half_t)(acc0 * inv);
    if (lane < HD_ - 32) dst[32 + lane] = (half_t)(acc1 * inv);
  }
}

// ---------------- K5: output projections 288 -> 512 -----------------------

__global__ void outproj_kernel(const half_t* __restrict__ attn,
                               const half_t* __restrict__ ow16,
                               const float* __restrict__ out_b,
                               float* __restrict__ out) {
  int wave = blockIdx.x * 8 + (threadIdx.x >> 5);   // 384 mtiles * 32 ntiles
  int lane = threadIdx.x & 31;
  int mt = wave >> 5, nt = wave & 31;
  int row0 = mt * 16;                               // 0..6143 = b*1536 + t'
  int b = row0 / TTOT_;
  int tt = row0 - b * TTOT_;
  int branch = tt >> 9;
  int t = tt & 511;
  int n0 = nt * 16;
  const half_t* A = attn + (size_t)row0 * HHD_;
  const half_t* W = ow16 + (size_t)(branch * D_ + n0) * HHD_;
  v8f acc = {};
#pragma unroll
  for (int kt = 0; kt < HHD_ / 32; ++kt) {          // 288 = 9 * 32 exactly
    v16h a = load_a16(A + kt * 32, HHD_, lane);
    v16h bm = load_bT16(W + kt * 32, HHD_, lane);
    acc = wmma16(a, bm, acc);
  }
  int col = n0 + (lane & 15);
  float bias = out_b[branch * D_ + col];
  float* obase = out + (size_t)branch * (B_ * T_ * D_) +
                 ((size_t)b * T_ + t) * D_ + col;
#pragma unroll
  for (int r = 0; r < 8; ++r) {
    int mloc = 8 * (lane >> 4) + r;
    obase[(size_t)mloc * D_] = acc[r] + bias;
  }
}

// ---------------- launch ---------------------------------------------------

extern "C" void kernel_launch(void* const* d_in, const int* in_sizes, int n_in,
                              void* d_out, int out_size, void* d_ws,
                              size_t ws_size, hipStream_t stream) {
  (void)in_sizes; (void)n_in; (void)out_size; (void)ws_size;
  const float* x1      = (const float*)d_in[0];
  const float* x2      = (const float*)d_in[1];
  const float* x3      = (const float*)d_in[2];
  const float* xf      = (const float*)d_in[3];
  const float* emb     = (const float*)d_in[4];
  const float* u_noise = (const float*)d_in[5];
  const float* adaln_w = (const float*)d_in[6];
  const float* adaln_b = (const float*)d_in[7];
  const float* xfn_w   = (const float*)d_in[8];
  const float* xfn_b   = (const float*)d_in[9];
  const float* q_w     = (const float*)d_in[10];
  const float* q_b     = (const float*)d_in[11];
  const float* k_w     = (const float*)d_in[12];
  const float* k_b     = (const float*)d_in[13];
  const float* v_w     = (const float*)d_in[14];
  const float* v_b     = (const float*)d_in[15];
  const float* out_w   = (const float*)d_in[16];
  const float* out_b   = (const float*)d_in[17];

  char* ws = (char*)d_ws;
  size_t off = 0;
  float*  mod  = (float*)(ws + off);  off += (size_t)4 * B_ * 2 * D_ * 4;
  half_t* xn   = (half_t*)(ws + off); off += (size_t)NTOK_ALL_ * D_ * 2;
  half_t* Qf   = (half_t*)(ws + off);
  size_t qbytes = (size_t)B_ * H_ * TTOT_ * KPAD_ * 2; off += qbytes;
  half_t* Kf   = (half_t*)(ws + off);
  size_t kbytes = (size_t)B_ * H_ * NH_ * KPAD_ * 2;   off += kbytes;
  float*  Vf   = (float*)(ws + off);  off += (size_t)B_ * H_ * NH_ * HD_ * 4;
  half_t* at   = (half_t*)(ws + off); off += (size_t)B_ * TTOT_ * HHD_ * 2;
  half_t* qw16 = (half_t*)(ws + off); off += (size_t)3 * HHD_ * D_ * 2;
  half_t* kw16 = (half_t*)(ws + off); off += (size_t)H_ * HD_ * D_ * 2;
  half_t* vw16 = (half_t*)(ws + off); off += (size_t)H_ * HD_ * D_ * 2;
  half_t* ow16 = (half_t*)(ws + off); off += (size_t)3 * D_ * HHD_ * 2;

  hipMemsetAsync(Qf, 0, qbytes, stream);   // zero K-dim padding (36..63)
  hipMemsetAsync(Kf, 0, kbytes, stream);

  const int nqw = 3 * HHD_ * D_;           // 442368
  const int nkw = H_ * HD_ * D_;           // 147456
  const int now = 3 * D_ * HHD_;           // 442368
  cvt_kernel<<<(nqw + 255) / 256, 256, 0, stream>>>(q_w, qw16, nqw);
  cvt_kernel<<<(nkw + 255) / 256, 256, 0, stream>>>(k_w, kw16, nkw);
  cvt_kernel<<<(nkw + 255) / 256, 256, 0, stream>>>(v_w, vw16, nkw);
  cvt_kernel<<<(now + 255) / 256, 256, 0, stream>>>(out_w, ow16, now);

  mod_kernel<<<64, 256, 0, stream>>>(emb, adaln_w, adaln_b, xfn_w, xfn_b, mod);
  adaln_kernel<<<NTOK_ALL_ / 8, 256, 0, stream>>>(x1, x2, x3, xf, mod, xn);
  qproj_kernel<<<(384 * 18) / 8, 256, 0, stream>>>(xn, qw16, q_b, Qf);
  kv_kernel<<<(1024 * 3) / 8, 256, 0, stream>>>(xn + (size_t)6144 * D_,
                                                kw16, k_b, vw16, v_b, Kf, Vf);
  attn_kernel<<<dim3(TTOT_ / 16, H_, B_), 256, 0, stream>>>(Qf, Kf, Vf,
                                                            u_noise, at);
  outproj_kernel<<<(384 * 32) / 8, 256, 0, stream>>>(at, ow16, out_b,
                                                     (float*)d_out);
}
